// MLPLinkDecoder_51591147160281
// MI455X (gfx1250) — matrix operand order repaired and checked by reading, compiled, not run on hardware
//
#include <hip/hip_runtime.h>
#include <hip/hip_bf16.h>

// MLP link decoder: h = z@W + b (WMMA bf16, fp32 accum), then per-edge
// gather-dot-sigmoid. h (100000x256 f32, ~98MB) is materialized in d_ws and
// stays L2-resident (192MB L2) for the gather phase.
//
// d_ws layout:
//   [0, 128KB)            : W converted to bf16, pre-swizzled into the
//                           v_wmma_f32_16x16x32_bf16 B-fragment layout
//   [128KB, 128KB+M*256*4): h (fp32, row-major)

typedef __attribute__((ext_vector_type(16))) __bf16       v16bf;
typedef __attribute__((ext_vector_type(8)))  float        v8f;
typedef __attribute__((ext_vector_type(8)))  unsigned int v8u;
typedef __attribute__((ext_vector_type(4)))  unsigned int v4u;

#define DDIM   256
#define KSTEPS 8    // 256 / 32
#define NTILES 16   // 256 / 16

__device__ __forceinline__ unsigned int pack2_bf16(float lo, float hi) {
  // round-to-nearest-even f32 -> bf16, packed (lo in [15:0], hi in [31:16])
  unsigned int xl = __float_as_uint(lo);
  unsigned int xh = __float_as_uint(hi);
  xl = (xl + 0x7FFFu + ((xl >> 16) & 1u)) >> 16;
  xh = (xh + 0x7FFFu + ((xh >> 16) & 1u)) >> 16;
  return (xl & 0xFFFFu) | (xh << 16);
}

// ---------------------------------------------------------------------------
// Kernel 1: convert W (f32 [K=256][N=256]) to bf16 in B-fragment order.
// Flat dword index = ((kt*16 + nt)*32 + lane)*8 + v, where, in contiguous k order (same as A)
// (32x16 bf16 B): lanes 0-15 hold K = kt*32 + 2v,2v+1 at column nt*16+(lane&15),
// lanes 16-31 hold K = kt*32 + 16 + 2v, 2v+1.
// ---------------------------------------------------------------------------
__global__ void wconv_kernel(const float* __restrict__ W,
                             unsigned int* __restrict__ wsw) {
  int tid  = blockIdx.x * blockDim.x + threadIdx.x;   // 32768 threads total
  int v    = tid & 7;
  int lane = (tid >> 3) & 31;
  int tile = tid >> 8;          // kt*16 + nt
  int nt   = tile & 15;
  int kt   = tile >> 4;
  int col  = nt * 16 + (lane & 15);
  int k    = kt * 32 + ((lane >> 4) << 4) + (v << 1);
  float lo = W[(size_t)k * DDIM + col];
  float hi = W[(size_t)(k + 1) * DDIM + col];
  wsw[tid] = pack2_bf16(lo, hi);
}

// ---------------------------------------------------------------------------
// Kernel 2: h = z @ W + b. One wave per 16-row strip, full 256 columns.
// Phase 1: convert the whole 16x256 A strip to bf16 fragments ONCE (64 VGPRs).
// Phase 2: nt outer / kt inner -> a single live v8f accumulator chained
// through 8 WMMAs, stored immediately. Keeps VGPR pressure ~150 (no spills).
// ---------------------------------------------------------------------------
__global__ void gemm_kernel(const float* __restrict__ z,
                            const unsigned int* __restrict__ wsw,
                            const float* __restrict__ bias,
                            float* __restrict__ h, int M) {
  const int lane    = threadIdx.x & 31;
  const int wave    = threadIdx.x >> 5;
  const int lanelo  = lane & 15;
  const int khalf   = lane >> 4;
  const int rowbase = (blockIdx.x * 8 + wave) * 16;
  if (rowbase >= M) return;     // wave-uniform: EXEC stays all-1s for WMMA

  // ---- Phase 1: load + convert all A fragments (contiguous k = 16*khalf + s, same as B) ----
  const float* zrow = z + (size_t)(rowbase + lanelo) * DDIM;
  v8u adw[KSTEPS];
  #pragma unroll
  for (int kt = 0; kt < KSTEPS; ++kt) {
    #pragma unroll
    for (int v = 0; v < 8; ++v) {
      // pair index: lanes0-15: {0..7}, lanes16-31: {8..15}
      int pair = v + (khalf << 3);
      int k    = kt * 32 + (pair << 1);
      float2 f = *(const float2*)(zrow + k);   // one 128B line per row/K-step
      adw[kt][v] = pack2_bf16(f.x, f.y);
    }
  }

  // ---- Phase 2: one column tile at a time, full K reduction, then store ----
  #pragma unroll 1
  for (int nt = 0; nt < NTILES; ++nt) {
    v8f acc = {};
    #pragma unroll
    for (int kt = 0; kt < KSTEPS; ++kt) {
      const v4u* bp = (const v4u*)wsw + ((size_t)((kt * 16 + nt) * 32 + lane) * 2);
      v4u b0 = bp[0];
      v4u b1 = bp[1];
      v8u bdw;
      bdw[0] = b0[0]; bdw[1] = b0[1]; bdw[2] = b0[2]; bdw[3] = b0[3];
      bdw[4] = b1[0]; bdw[5] = b1[1]; bdw[6] = b1[2]; bdw[7] = b1[3];
      v16bf a = __builtin_bit_cast(v16bf, adw[kt]);
      v16bf b = __builtin_bit_cast(v16bf, bdw);
      acc = __builtin_amdgcn_wmma_f32_16x16x32_bf16(
          /*neg_a=*/false, a, /*neg_b=*/false, b,
          /*c_mod=*/(short)0, acc, /*reuse_a=*/false, /*reuse_b=*/false);
    }

    // epilogue: add bias, store h (C/D layout: VGPR r -> M = r + 8*khalf)
    int col  = nt * 16 + lanelo;
    float bv = bias[col];
    #pragma unroll
    for (int r = 0; r < 8; ++r) {
      int row = rowbase + (khalf << 3) + r;
      h[(size_t)row * DDIM + col] = acc[r] + bv;
    }
  }
}

// ---------------------------------------------------------------------------
// Kernel 3: per-edge dot + sigmoid. One wave32 per edge: each lane owns 8
// floats (2x float4) of each endpoint row; shfl_xor reduction; lane 0 writes.
// ---------------------------------------------------------------------------
__global__ void edge_kernel(const float* __restrict__ h,
                            const long long* __restrict__ ei,
                            float* __restrict__ out, int E) {
  int gid  = blockIdx.x * blockDim.x + threadIdx.x;
  int e    = gid >> 5;
  int lane = gid & 31;
  if (e >= E) return;

  int s = (int)ei[e];
  int d = (int)ei[(size_t)E + e];

  const float4* hs = (const float4*)(h + (size_t)s * DDIM) + lane * 2;
  const float4* hd = (const float4*)(h + (size_t)d * DDIM) + lane * 2;
  float4 a0 = hs[0], a1 = hs[1];
  float4 b0 = hd[0], b1 = hd[1];

  float p = a0.x * b0.x;
  p = fmaf(a0.y, b0.y, p);
  p = fmaf(a0.z, b0.z, p);
  p = fmaf(a0.w, b0.w, p);
  p = fmaf(a1.x, b1.x, p);
  p = fmaf(a1.y, b1.y, p);
  p = fmaf(a1.z, b1.z, p);
  p = fmaf(a1.w, b1.w, p);

  #pragma unroll
  for (int off = 16; off >= 1; off >>= 1)
    p += __shfl_xor(p, off, 32);

  if (lane == 0)
    out[e] = 1.0f / (1.0f + __expf(-p));
}

// ---------------------------------------------------------------------------
extern "C" void kernel_launch(void* const* d_in, const int* in_sizes, int n_in,
                              void* d_out, int out_size, void* d_ws, size_t ws_size,
                              hipStream_t stream) {
  const float*     z  = (const float*)d_in[0];
  const long long* ei = (const long long*)d_in[1];   // int64 per reference
  const float*     W  = (const float*)d_in[2];
  const float*     b  = (const float*)d_in[3];
  float*           out = (float*)d_out;

  int M = in_sizes[0] / DDIM;   // 100000 nodes
  int E = in_sizes[1] / 2;      // 300000 edges

  unsigned int* wsw = (unsigned int*)d_ws;                        // 128 KB
  float* h = (float*)((char*)d_ws + (size_t)DDIM * DDIM * 2);     // M*256*4 B

  // 1) W -> bf16, B-fragment swizzled (32768 dwords)
  wconv_kernel<<<(DDIM * DDIM / 2 + 255) / 256, 256, 0, stream>>>(W, wsw);

  // 2) GEMM: one wave per 16-row strip, 8 waves per block
  int strips = (M + 15) / 16;
  gemm_kernel<<<(strips + 7) / 8, 256, 0, stream>>>(z, wsw, b, h, M);

  // 3) edges: one wave per edge, 8 waves per block
  edge_kernel<<<(E + 7) / 8, 256, 0, stream>>>(h, ei, out, E);
}
